// BHLinear_38465727103159
// MI455X (gfx1250) — matrix operand order repaired and checked by reading, not compiled, not run on hardware
//
#include <hip/hip_runtime.h>
#include <hip/hip_bf16.h>

typedef __bf16 bf16;
typedef __attribute__((ext_vector_type(16))) __bf16 v16bf;
typedef __attribute__((ext_vector_type(8)))  __bf16 v8bf;
typedef __attribute__((ext_vector_type(8)))  float  v8f;

#define DIM     4096
#define ZSTR    4104          // padded LDS row stride (bf16 elems): 8208 B, 16B aligned
#define ROWS    16            // batch rows per workgroup
#define NROWS   16384         // 4 * 4096

// workspace layout (bf16 element offsets)
#define WQ_ELEMS (2*64*4*2*32*16)   // 524288  : W[d][g] = inner_B[d][g] @ (H/8), swizzled
#define FQ_ELEMS (128*2*32*16)      // 131072  : final_B swizzled
#define HQ_ELEMS (4*2*32*16)        // 4096    : H/8 swizzled
#define FQ_OFF   WQ_ELEMS
#define HQ_OFF   (WQ_ELEMS + FQ_ELEMS)

// ---------------- prep kernels: build swizzled bf16 B-operand tables ----------------

// Wq index = ((((d*64+g)*4 + nt)*2 + ks)*32 + lane)*16 + e
__global__ void __launch_bounds__(256) prep_w(const float* __restrict__ innerB,
                                              bf16* __restrict__ wq) {
    int idx = blockIdx.x * 256 + threadIdx.x;          // 524288 threads
    int e    = idx & 15;
    int lane = (idx >> 4) & 31;
    int ks   = (idx >> 9) & 1;
    int nt   = (idx >> 10) & 3;
    int g    = (idx >> 12) & 63;
    int d    = (idx >> 18) & 1;
    int k = ks * 32 + ((e >> 3) << 4) + (lane >> 4) * 8 + (e & 7);   // K index 0..63
    int i = nt * 16 + (lane & 15);                     // output column 0..63
    const float* Bg = innerB + ((size_t)((d * 64 + g) * 64 + k)) * 64;
    float acc = 0.f;
    for (int j = 0; j < 64; ++j) {
        float s = (__popc(j & i) & 1) ? -0.125f : 0.125f;   // (H/8)[j][i]
        acc += Bg[j] * s;
    }
    wq[idx] = (bf16)acc;
}

// Fq index = ((r*2 + nt)*32 + lane)*16 + e
__global__ void __launch_bounds__(256) prep_f(const float* __restrict__ finalB,
                                              bf16* __restrict__ fq) {
    int idx = blockIdx.x * 256 + threadIdx.x;          // 131072 threads
    int e    = idx & 15;
    int lane = (idx >> 4) & 31;
    int nt   = (idx >> 9) & 1;
    int r    = (idx >> 10) & 127;
    int k = ((e >> 3) << 4) + (lane >> 4) * 8 + (e & 7);   // 0..31
    int n = nt * 16 + (lane & 15);                     // 0..31
    fq[idx] = (bf16)finalB[(size_t)(r * 32 + k) * 32 + n];
}

// Hq index = ((nt*2 + ks)*32 + lane)*16 + e
__global__ void __launch_bounds__(256) prep_h(bf16* __restrict__ hq) {
    int idx = blockIdx.x * 256 + threadIdx.x;          // 4096 threads
    int e    = idx & 15;
    int lane = (idx >> 4) & 31;
    int ks   = (idx >> 9) & 1;
    int nt   = (idx >> 10) & 3;
    int k = ks * 32 + ((e >> 3) << 4) + (lane >> 4) * 8 + (e & 7);
    int n = nt * 16 + (lane & 15);
    hq[idx] = (bf16)((__popc(k & n) & 1) ? -0.125f : 0.125f);
}

// ---------------- fused main kernel ----------------

__device__ __forceinline__ v8f wmma_bf16(v16bf a, v16bf b, v8f c) {
    return __builtin_amdgcn_wmma_f32_16x16x32_bf16(false, a, false, b, (short)0, c,
                                                   false, false);
}

// left-Hadamard pass over all 64-group axis, in place in LDS
__device__ __forceinline__ void phaseB(bf16* Z, const bf16* __restrict__ hq,
                                       int w, int lane) {
    const int nl = lane & 15;
    const int kh = lane >> 4;
    v16bf hb[4][2];
#pragma unroll
    for (int nt = 0; nt < 4; ++nt)
#pragma unroll
        for (int s = 0; s < 2; ++s)
            hb[nt][s] = ((const v16bf*)hq)[(nt * 2 + s) * 32 + lane];

    for (int t = 0; t < 8; ++t) {
        int i = w * 8 + t;                              // this wave's column set
        v16bf a[2];
#pragma unroll
        for (int s = 0; s < 2; ++s)
#pragma unroll
            for (int e = 0; e < 8; ++e) {
                int g0 = s * 32 + kh * 8 + e;           // K (= group) indices
                int g1 = s * 32 + 16 + kh * 8 + e;
                a[s][e]     = Z[nl * ZSTR + g0 * 64 + i];
                a[s][e + 8] = Z[nl * ZSTR + g1 * 64 + i];
            }
        v8f acc[4] = {};
#pragma unroll
        for (int nt = 0; nt < 4; ++nt)
#pragma unroll
            for (int s = 0; s < 2; ++s)
                acc[nt] = wmma_bf16(a[s], hb[nt][s], acc[nt]);
#pragma unroll
        for (int nt = 0; nt < 4; ++nt)
#pragma unroll
            for (int v = 0; v < 8; ++v)
                Z[(v + 8 * kh) * ZSTR + (nt * 16 + nl) * 64 + i] = (bf16)acc[nt][v];
    }
}

__global__ void __launch_bounds__(256) bhl_main(const float* __restrict__ x,
                                                const bf16* __restrict__ wq,
                                                const bf16* __restrict__ fq,
                                                const bf16* __restrict__ hq,
                                                float* __restrict__ out) {
    extern __shared__ bf16 Z[];                         // ROWS * ZSTR bf16
    const int tid  = threadIdx.x;
    const int lane = tid & 31;
    const int w    = tid >> 5;                          // wave 0..7
    const int nl   = lane & 15;
    const int kh   = lane >> 4;
    const size_t row0 = (size_t)blockIdx.x * ROWS;

    // ---- depth 0, phase A: Z = (x_g @ W0_g) per group, A from global f32 ----
    for (int gi = 0; gi < 8; ++gi) {
        int g = w * 8 + gi;
        v16bf a[2];
#pragma unroll
        for (int s = 0; s < 2; ++s) {
            const float* p = x + (row0 + nl) * DIM + g * 64 + s * 32 + kh * 8;
            float4 f0 = *(const float4*)(p);
            float4 f1 = *(const float4*)(p + 4);
            float4 f2 = *(const float4*)(p + 16);
            float4 f3 = *(const float4*)(p + 20);
            a[s][0] = (bf16)f0.x;  a[s][1] = (bf16)f0.y;
            a[s][2] = (bf16)f0.z;  a[s][3] = (bf16)f0.w;
            a[s][4] = (bf16)f1.x;  a[s][5] = (bf16)f1.y;
            a[s][6] = (bf16)f1.z;  a[s][7] = (bf16)f1.w;
            a[s][8]  = (bf16)f2.x; a[s][9]  = (bf16)f2.y;
            a[s][10] = (bf16)f2.z; a[s][11] = (bf16)f2.w;
            a[s][12] = (bf16)f3.x; a[s][13] = (bf16)f3.y;
            a[s][14] = (bf16)f3.z; a[s][15] = (bf16)f3.w;
        }
        v8f acc[4] = {};
#pragma unroll
        for (int nt = 0; nt < 4; ++nt)
#pragma unroll
            for (int s = 0; s < 2; ++s) {
                v16bf b = ((const v16bf*)wq)[(((0 * 64 + g) * 4 + nt) * 2 + s) * 32 + lane];
                acc[nt] = wmma_bf16(a[s], b, acc[nt]);
            }
#pragma unroll
        for (int nt = 0; nt < 4; ++nt)
#pragma unroll
            for (int v = 0; v < 8; ++v)
                Z[(v + 8 * kh) * ZSTR + g * 64 + nt * 16 + nl] = (bf16)acc[nt][v];
    }
    __syncthreads();
    phaseB(Z, hq, w, lane);                             // depth 0 left-Hadamard
    __syncthreads();

    // ---- depth 1, phase A: Z_g = (Z_g @ W1_g), A from LDS ----
    for (int gi = 0; gi < 8; ++gi) {
        int g = w * 8 + gi;
        v16bf a[2];
#pragma unroll
        for (int s = 0; s < 2; ++s) {
            v8bf c0 = *(const v8bf*)(Z + nl * ZSTR + g * 64 + s * 32 + kh * 8);
            v8bf c1 = *(const v8bf*)(Z + nl * ZSTR + g * 64 + s * 32 + 16 + kh * 8);
#pragma unroll
            for (int e = 0; e < 8; ++e) { a[s][e] = c0[e]; a[s][e + 8] = c1[e]; }
        }
        v8f acc[4] = {};
#pragma unroll
        for (int nt = 0; nt < 4; ++nt)
#pragma unroll
            for (int s = 0; s < 2; ++s) {
                v16bf b = ((const v16bf*)wq)[(((1 * 64 + g) * 4 + nt) * 2 + s) * 32 + lane];
                acc[nt] = wmma_bf16(a[s], b, acc[nt]);
            }
#pragma unroll
        for (int nt = 0; nt < 4; ++nt)
#pragma unroll
            for (int v = 0; v < 8; ++v)
                Z[(v + 8 * kh) * ZSTR + g * 64 + nt * 16 + nl] = (bf16)acc[nt][v];
    }
    __syncthreads();
    phaseB(Z, hq, w, lane);                             // depth 1 left-Hadamard
    __syncthreads();

    // ---- final: out_r = Z_r(16x32) @ final_B_r(32x32), f32 stores ----
    for (int t = 0; t < 16; ++t) {
        int r = w * 16 + t;
        v16bf a;
        v8bf c0 = *(const v8bf*)(Z + nl * ZSTR + r * 32 + kh * 8);
        v8bf c1 = *(const v8bf*)(Z + nl * ZSTR + r * 32 + 16 + kh * 8);
#pragma unroll
        for (int e = 0; e < 8; ++e) { a[e] = c0[e]; a[e + 8] = c1[e]; }
        v8f acc[2] = {};
#pragma unroll
        for (int nt = 0; nt < 2; ++nt) {
            v16bf b = ((const v16bf*)fq)[(r * 2 + nt) * 32 + lane];
            acc[nt] = wmma_bf16(a, b, acc[nt]);
        }
#pragma unroll
        for (int nt = 0; nt < 2; ++nt)
#pragma unroll
            for (int v = 0; v < 8; ++v)
                out[(row0 + v + 8 * kh) * DIM + r * 32 + nt * 16 + nl] = acc[nt][v];
    }
}

// ---------------- launcher ----------------

extern "C" void kernel_launch(void* const* d_in, const int* in_sizes, int n_in,
                              void* d_out, int out_size, void* d_ws, size_t ws_size,
                              hipStream_t stream) {
    (void)in_sizes; (void)n_in; (void)out_size; (void)ws_size;
    const float* x      = (const float*)d_in[0];
    const float* innerB = (const float*)d_in[1];
    const float* finalB = (const float*)d_in[2];
    float* out = (float*)d_out;
    bf16*  ws  = (bf16*)d_ws;
    bf16*  wq  = ws;
    bf16*  fq  = ws + FQ_OFF;
    bf16*  hq  = ws + HQ_OFF;

    prep_w<<<WQ_ELEMS / 256, 256, 0, stream>>>(innerB, wq);
    prep_f<<<FQ_ELEMS / 256, 256, 0, stream>>>(finalB, fq);
    prep_h<<<HQ_ELEMS / 256, 256, 0, stream>>>(hq);

    size_t shmem = (size_t)ROWS * ZSTR * sizeof(bf16);  // 131328 B < 320 KB WGP LDS
    (void)hipFuncSetAttribute((const void*)bhl_main,
                              hipFuncAttributeMaxDynamicSharedMemorySize, (int)shmem);
    bhl_main<<<NROWS / ROWS, 256, shmem, stream>>>(x, wq, fq, hq, out);
}
